// Fusion_38414187495561
// MI455X (gfx1250) — matrix operand order repaired and checked by reading, not compiled, not run on hardware
//
#include <hip/hip_runtime.h>
#include <math.h>

#define DIMC 128
#define SZ   64
#define NFQ  8
#define NB   16
#define C2Q  65
#define FSQ  33
#define HIDQ 32
#define HWQ  (SZ*SZ)       /* 4096 */
#define SFSQ (SZ*FSQ)      /* 2112 */
#define EPSQ 1e-5f
#define PIF  3.14159265358979323846f

typedef _Float16 half_t;
typedef __attribute__((ext_vector_type(16))) _Float16 v16h;
typedef __attribute__((ext_vector_type(8)))  _Float16 v8h;
typedef __attribute__((ext_vector_type(8)))  float    v8f;

// ---------------------------------------------------------------------------
// K1: BatchNorm both streams, res = r + t, pooled mean of (rn+tn)
// grid = B*DIM blocks (one (b,c) plane each), 256 threads
// ---------------------------------------------------------------------------
__global__ void k_bn_res_pool(const float* __restrict__ r, const float* __restrict__ t,
                              const float* __restrict__ gr, const float* __restrict__ br_,
                              const float* __restrict__ mr, const float* __restrict__ vr,
                              const float* __restrict__ gt, const float* __restrict__ bt_,
                              const float* __restrict__ mt, const float* __restrict__ vt,
                              float* __restrict__ rn, float* __restrict__ tn,
                              float* __restrict__ res, float* __restrict__ pooled)
{
    int bc = blockIdx.x;
    int c  = bc % DIMC;
    float sr = gr[c] * rsqrtf(vr[c] + EPSQ);
    float st = gt[c] * rsqrtf(vt[c] + EPSQ);
    float mur = mr[c], mut = mt[c], ber = br_[c], bet = bt_[c];
    long base = (long)bc * HWQ;
    float acc = 0.f;
    for (int i = threadIdx.x; i < HWQ; i += 256) {
        float rv = r[base + i], tv = t[base + i];
        float rnv = (rv - mur) * sr + ber;
        float tnv = (tv - mut) * st + bet;
        rn[base + i] = rnv;
        tn[base + i] = tnv;
        res[base + i] = rv + tv;
        acc += rnv + tnv;
    }
    __shared__ float sred[256];
    sred[threadIdx.x] = acc;
    __syncthreads();
    for (int s = 128; s > 0; s >>= 1) {
        if (threadIdx.x < s) sred[threadIdx.x] += sred[threadIdx.x + s];
        __syncthreads();
    }
    if (threadIdx.x == 0) pooled[bc] = sred[0] * (1.f / (float)HWQ);
}

// ---------------------------------------------------------------------------
// K2: routing MLP: pooled@fc1^T -> StarReLU -> @fc2^T -> softmax over NF
// grid = B blocks, 128 threads (one per channel)
// ---------------------------------------------------------------------------
__global__ void k_routing(const float* __restrict__ pooled, const float* __restrict__ fc1,
                          const float* __restrict__ sp, const float* __restrict__ bp,
                          const float* __restrict__ fc2, float* __restrict__ route)
{
    int b = blockIdx.x;
    int t = threadIdx.x;   // 0..127
    __shared__ float p[DIMC];
    __shared__ float h[HIDQ];
    p[t] = pooled[b * DIMC + t];
    __syncthreads();
    if (t < HIDQ) {
        float a = 0.f;
        for (int c = 0; c < DIMC; c++) a += fc1[t * DIMC + c] * p[c];
        float rl = fmaxf(a, 0.f);
        h[t] = sp[0] * rl * rl + bp[0];
    }
    __syncthreads();
    float lg[NFQ];
    float mx = -1e30f;
    for (int f = 0; f < NFQ; f++) {
        int row = f * DIMC + t;
        float a = 0.f;
        for (int k = 0; k < HIDQ; k++) a += fc2[row * HIDQ + k] * h[k];
        lg[f] = a;
        mx = fmaxf(mx, a);
    }
    float den = 0.f;
    for (int f = 0; f < NFQ; f++) { lg[f] = expf(lg[f] - mx); den += lg[f]; }
    float inv = 1.f / den;
    for (int f = 0; f < NFQ; f++) route[(b * NFQ + f) * DIMC + t] = lg[f] * inv;
}

// ---------------------------------------------------------------------------
// K3: WMMA GEMM:  Y[b][m][n] = act( sum_k W[m][k] * X[b][k][n] + bias[m] )
// M,K compile-time (128x128 or 65x65) -> padding masks constant-fold.
// Block computes 32(M) x 64(N). A tile 32 x Kpad staged once (row-major f16,
// fragment runs are 16B -> ds_load_b128). X tile staged per K-step in
// column-major [64 cols][40 halves] so each lane's B fragment is two
// contiguous 16B LDS reads. All hot-path addressing is 32-bit.
// act: 0=none, 1=leaky(0.1), 2=StarReLU(sp,bp)
// grid = (ld/64, ceil(M/32), B), block = 128 (4 waves, 2 wmma each / K-step)
// ---------------------------------------------------------------------------
template <int M, int K>
__global__ void k_wmma_gemm(const float* __restrict__ W, const float* __restrict__ X,
                            const float* __restrict__ bias, float* __restrict__ Y,
                            int ld, long xbs, long ybs,
                            int act, const float* __restrict__ sp, const float* __restrict__ bp)
{
    constexpr int KP = (K + 31) & ~31;
    constexpr int MT = 32;
    constexpr int XST = 40;                       // halves per column (padded)
    __shared__ __align__(16) half_t ash[MT * KP];
    __shared__ __align__(16) half_t xsh[64 * XST];

    int tid  = threadIdx.x;
    int lane = tid & 31;
    int wave = tid >> 5;
    int m0 = blockIdx.y * MT;
    int nb = blockIdx.x * 64;
    const float* Xb = X + (long)blockIdx.z * xbs;
    float*       Yb = Y + (long)blockIdx.z * ybs;

    // stage A tile (MT x KP), zero-padded, unrolled & branch-free
    #pragma unroll
    for (int g = 0; g < (MT * KP) / 128; g++) {
        int i  = g * 128 + tid;
        int mr = i / KP, kk = i % KP;
        int m  = m0 + mr;
        int sm = (m  < M) ? m  : (M - 1);
        int sk = (kk < K) ? kk : (K - 1);
        float v = W[sm * K + sk];
        ash[i] = (half_t)((m < M && kk < K) ? v : 0.f);
    }

    int lm = lane & 15;
    int hi = lane >> 4;
    int ncol = nb + wave * 16 + lm;

    v8f acc0 = {0.f,0.f,0.f,0.f,0.f,0.f,0.f,0.f};
    v8f acc1 = {0.f,0.f,0.f,0.f,0.f,0.f,0.f,0.f};
    #pragma unroll
    for (int ks = 0; ks < KP / 32; ks++) {
        int kb = ks * 32;
        __syncthreads();
        // stage X tile rows kb..kb+31, cols nb..nb+63 into column-major LDS
        #pragma unroll
        for (int g = 0; g < 16; g++) {
            int i = g * 128 + tid;
            int row = i >> 6, col = i & 63;
            int k  = kb + row;
            int sk = (k < K) ? k : (K - 1);
            float v = Xb[sk * ld + nb + col];          // 32-bit indexing
            xsh[col * XST + row] = (half_t)((k < K) ? v : 0.f);
        }
        __syncthreads();
        v16h a0, a1;
        // A fragments: two 16B runs per fragment (ds_load_b128)
        #pragma unroll
        for (int v = 0; v < 8; v++) {
            int kk = kb + 2 * v + (hi ? 16 : 0);
            a0[2 * v]     = ash[lm * KP + kk];
            a0[2 * v + 1] = ash[lm * KP + kk + 1];
            a1[2 * v]     = ash[(16 + lm) * KP + kk];
            a1[2 * v + 1] = ash[(16 + lm) * KP + kk + 1];
        }
        // B fragment: 16 consecutive halves at [col][hi*16] -> 2x ds_load_b128
        union { v16h v; v8h h[2]; } bu;
        const v8h* bp16 = (const v8h*)(xsh + (wave * 16 + lm) * XST + hi * 16);
        bu.h[0] = bp16[0];
        bu.h[1] = bp16[1];
        acc0 = __builtin_amdgcn_wmma_f32_16x16x32_f16(false, a0, false, bu.v,
                                                      (short)0, acc0, false, false);
        acc1 = __builtin_amdgcn_wmma_f32_16x16x32_f16(false, a1, false, bu.v,
                                                      (short)0, acc1, false, false);
    }

    float ss = (act == 2) ? sp[0] : 0.f;
    float sb = (act == 2) ? bp[0] : 0.f;
    #pragma unroll
    for (int i = 0; i < 8; i++) {
        int mA = m0 + i + (hi ? 8 : 0);
        int mB = mA + 16;
        if (mA < M) {
            float v = acc0[i] + bias[mA];
            if (act == 1)      v = (v > 0.f) ? v : 0.1f * v;
            else if (act == 2) { float rl = fmaxf(v, 0.f); v = ss * rl * rl + sb; }
            Yb[mA * ld + ncol] = v;
        }
        if (mB < M) {
            float v = acc1[i] + bias[mB];
            if (act == 1)      v = (v > 0.f) ? v : 0.1f * v;
            else if (act == 2) { float rl = fmaxf(v, 0.f); v = ss * rl * rl + sb; }
            Yb[mB * ld + ncol] = v;
        }
    }
}

// ---------------------------------------------------------------------------
// K4: spatial rfft2 (ortho) of a [B,DIM,64,64] plane -> amp/phase [B,DIM,64,33]
// grid = B*DIM blocks, 256 threads; separable DFT with 64-entry twiddle table
// ---------------------------------------------------------------------------
__global__ void k_rfft2_amp_phase(const float* __restrict__ X,
                                  float* __restrict__ amp, float* __restrict__ ph)
{
    __shared__ float xs[HWQ];
    __shared__ float fre[SFSQ], fim[SFSQ];
    __shared__ float twc[SZ], tws[SZ];
    int t = threadIdx.x;
    long base = (long)blockIdx.x * HWQ;
    for (int i = t; i < HWQ; i += 256) xs[i] = X[base + i];
    if (t < SZ) { float a = -2.f * PIF * (float)t / (float)SZ; twc[t] = cosf(a); tws[t] = sinf(a); }
    __syncthreads();
    // stage 1: row DFT (x -> u)
    for (int idx = t; idx < SFSQ; idx += 256) {
        int y = idx / FSQ, u = idx % FSQ;
        float re = 0.f, im = 0.f;
        for (int x = 0; x < SZ; x++) {
            int k = (u * x) & 63;
            float v = xs[y * SZ + x];
            re += v * twc[k];
            im += v * tws[k];
        }
        fre[idx] = re; fim[idx] = im;
    }
    __syncthreads();
    // stage 2: column DFT (y -> v), ortho 1/64, then amp/phase
    long ob = (long)blockIdx.x * SFSQ;
    for (int idx = t; idx < SFSQ; idx += 256) {
        int v = idx / FSQ, u = idx % FSQ;
        float re = 0.f, im = 0.f;
        for (int y = 0; y < SZ; y++) {
            int k = (v * y) & 63;
            float cr = twc[k], si = tws[k];
            float ar = fre[y * FSQ + u], ai = fim[y * FSQ + u];
            re += ar * cr - ai * si;
            im += ar * si + ai * cr;
        }
        re *= (1.f / 64.f); im *= (1.f / 64.f);
        amp[ob + idx] = sqrtf(re * re + im * im);
        ph[ob + idx]  = atan2f(im, re);
    }
}

// ---------------------------------------------------------------------------
// K5: channel rfft (128 -> 65, ortho): A [B,128,2112] -> abs/angle [B,65,2112]
// grid = (2112/64, B), 64 threads; LDS transpose tile for coalescing
// ---------------------------------------------------------------------------
__global__ void k_chan_rfft(const float* __restrict__ A,
                            float* __restrict__ fa, float* __restrict__ fp)
{
    __shared__ float col[DIMC][SZ + 1];
    __shared__ float twc[DIMC], tws[DIMC];
    int b = blockIdx.y;
    int s0 = blockIdx.x * 64;
    int t = threadIdx.x;  // 0..63
    const float* Ab = A + (long)b * DIMC * SFSQ;
    for (int c = 0; c < DIMC; c++) col[c][t] = Ab[(long)c * SFSQ + s0 + t];
    {
        float a0 = -2.f * PIF * (float)t / 128.f;
        float a1 = -2.f * PIF * (float)(t + 64) / 128.f;
        twc[t] = cosf(a0); tws[t] = sinf(a0);
        twc[t + 64] = cosf(a1); tws[t + 64] = sinf(a1);
    }
    __syncthreads();
    float* oa = fa + (long)b * C2Q * SFSQ;
    float* op = fp + (long)b * C2Q * SFSQ;
    const float nrm = 0.08838834764831845f;   // 1/sqrt(128)
    for (int k = 0; k < C2Q; k++) {
        float re = 0.f, im = 0.f;
        for (int c = 0; c < DIMC; c++) {
            int idx = (k * c) & 127;
            float v = col[c][t];
            re += v * twc[idx];
            im += v * tws[idx];
        }
        re *= nrm; im *= nrm;
        oa[(long)k * SFSQ + s0 + t] = sqrtf(re * re + im * im);
        op[(long)k * SFSQ + s0 + t] = atan2f(im, re);
    }
}

// ---------------------------------------------------------------------------
// K7: comp = ae*e^{i pe}; irfft(n=128, ortho) along channel; multiply by a_orig
// grid = (2112/64, B), 64 threads
// ---------------------------------------------------------------------------
__global__ void k_chan_irfft_mul(const float* __restrict__ ae, const float* __restrict__ pe,
                                 const float* __restrict__ aorig, float* __restrict__ out)
{
    __shared__ float fre[C2Q][SZ], fim[C2Q][SZ];
    __shared__ float twc[DIMC], tws[DIMC];
    int b = blockIdx.y;
    int s0 = blockIdx.x * 64;
    int t = threadIdx.x;
    const float* aeb = ae + (long)b * C2Q * SFSQ;
    const float* peb = pe + (long)b * C2Q * SFSQ;
    for (int k = 0; k < C2Q; k++) {
        float av = aeb[(long)k * SFSQ + s0 + t];
        float pv = peb[(long)k * SFSQ + s0 + t];
        fre[k][t] = av * cosf(pv);
        fim[k][t] = av * sinf(pv);
    }
    {
        float a0 = 2.f * PIF * (float)t / 128.f;
        float a1 = 2.f * PIF * (float)(t + 64) / 128.f;
        twc[t] = cosf(a0); tws[t] = sinf(a0);
        twc[t + 64] = cosf(a1); tws[t + 64] = sinf(a1);
    }
    __syncthreads();
    const float* ab = aorig + (long)b * DIMC * SFSQ;
    float* ob = out + (long)b * DIMC * SFSQ;
    const float nrm = 0.08838834764831845f;   // 1/sqrt(128)
    for (int c = 0; c < DIMC; c++) {
        float acc = fre[0][t] + ((c & 1) ? -fre[64][t] : fre[64][t]);
        for (int k = 1; k < 64; k++) {
            int idx = (k * c) & 127;
            acc += 2.f * (fre[k][t] * twc[idx] - fim[k][t] * tws[idx]);
        }
        acc *= nrm;
        ob[(long)c * SFSQ + s0 + t] = acc * ab[(long)c * SFSQ + s0 + t];
    }
}

// ---------------------------------------------------------------------------
// K9: rt_s = rsc*e^{i rspe} + tsc*e^{i tspe}; *= routing-weighted complex filter;
//     irfft2 (ortho); + res  -> out1.  grid = B*DIM, 256 threads
// ---------------------------------------------------------------------------
__global__ void k_combine_irfft2(const float* __restrict__ rsc, const float* __restrict__ rspe,
                                 const float* __restrict__ tsc, const float* __restrict__ tspe,
                                 const float* __restrict__ route, const float* __restrict__ cw,
                                 const float* __restrict__ res, float* __restrict__ out1)
{
    __shared__ float Gre[SFSQ], Gim[SFSQ];
    __shared__ float g2re[SFSQ], g2im[SFSQ];
    __shared__ float twc[SZ], tws[SZ];
    __shared__ float rf[NFQ];
    int bc = blockIdx.x;
    int b = bc / DIMC, c = bc % DIMC;
    int t = threadIdx.x;
    if (t < SZ) { float a = 2.f * PIF * (float)t / (float)SZ; twc[t] = cosf(a); tws[t] = sinf(a); }
    if (t < NFQ) rf[t] = route[(b * NFQ + t) * DIMC + c];
    __syncthreads();
    long base = (long)bc * SFSQ;
    for (int idx = t; idx < SFSQ; idx += 256) {
        int h = idx / FSQ, u = idx % FSQ;
        float a1 = rsc[base + idx], p1 = rspe[base + idx];
        float a2 = tsc[base + idx], p2 = tspe[base + idx];
        float sre = a1 * cosf(p1) + a2 * cosf(p2);
        float sim = a1 * sinf(p1) + a2 * sinf(p2);
        const float* cwp = cw + ((long)(h * FSQ + u) * NFQ) * 2;
        float wre = 0.f, wim = 0.f;
        #pragma unroll
        for (int f = 0; f < NFQ; f++) { wre += rf[f] * cwp[2 * f]; wim += rf[f] * cwp[2 * f + 1]; }
        Gre[idx] = sre * wre - sim * wim;
        Gim[idx] = sre * wim + sim * wre;
    }
    __syncthreads();
    // inverse column DFT (v -> y)
    for (int idx = t; idx < SFSQ; idx += 256) {
        int y = idx / FSQ, u = idx % FSQ;
        float re = 0.f, im = 0.f;
        for (int v = 0; v < SZ; v++) {
            int k = (v * y) & 63;
            float cr = twc[k], si = tws[k];
            float gr = Gre[v * FSQ + u], gi = Gim[v * FSQ + u];
            re += gr * cr - gi * si;
            im += gr * si + gi * cr;
        }
        g2re[idx] = re; g2im[idx] = im;
    }
    __syncthreads();
    // inverse row rDFT with Hermitian reconstruction (u -> x), ortho 1/64
    long ob = (long)bc * HWQ;
    for (int idx = t; idx < HWQ; idx += 256) {
        int y = idx / SZ, x = idx % SZ;
        float acc = g2re[y * FSQ] + ((x & 1) ? -g2re[y * FSQ + 32] : g2re[y * FSQ + 32]);
        for (int u = 1; u < 32; u++) {
            int k = (u * x) & 63;
            acc += 2.f * (g2re[y * FSQ + u] * twc[k] - g2im[y * FSQ + u] * tws[k]);
        }
        acc *= (1.f / 64.f);
        out1[ob + idx] = acc + res[ob + idx];
    }
}

// ---------------------------------------------------------------------------
// K10: FFN: BN -> depthwise 3x3 (SAME, no bias, correlation) -> + out1
// grid = B*DIM, 256 threads
// ---------------------------------------------------------------------------
__global__ void k_ffn(const float* __restrict__ out1,
                      const float* __restrict__ g, const float* __restrict__ be,
                      const float* __restrict__ mu, const float* __restrict__ var,
                      const float* __restrict__ dww, float* __restrict__ out)
{
    __shared__ float yt[SZ + 2][SZ + 2];
    int bc = blockIdx.x;
    int c = bc % DIMC;
    float sc = g[c] * rsqrtf(var[c] + EPSQ);
    float mv = mu[c], bb = be[c];
    const float* ip = out1 + (long)bc * HWQ;
    int t = threadIdx.x;
    for (int i = t; i < (SZ + 2) * (SZ + 2); i += 256) {
        int yy = i / (SZ + 2) - 1, xx = i % (SZ + 2) - 1;
        float v = 0.f;
        if (yy >= 0 && yy < SZ && xx >= 0 && xx < SZ) v = (ip[yy * SZ + xx] - mv) * sc + bb;
        yt[i / (SZ + 2)][i % (SZ + 2)] = v;
    }
    __syncthreads();
    float w0 = dww[c * 9 + 0], w1 = dww[c * 9 + 1], w2 = dww[c * 9 + 2];
    float w3 = dww[c * 9 + 3], w4 = dww[c * 9 + 4], w5 = dww[c * 9 + 5];
    float w6 = dww[c * 9 + 6], w7 = dww[c * 9 + 7], w8 = dww[c * 9 + 8];
    float* op = out + (long)bc * HWQ;
    for (int i = t; i < HWQ; i += 256) {
        int yy = i / SZ, xx = i % SZ;
        float s = yt[yy][xx] * w0 + yt[yy][xx + 1] * w1 + yt[yy][xx + 2] * w2
                + yt[yy + 1][xx] * w3 + yt[yy + 1][xx + 1] * w4 + yt[yy + 1][xx + 2] * w5
                + yt[yy + 2][xx] * w6 + yt[yy + 2][xx + 1] * w7 + yt[yy + 2][xx + 2] * w8;
        op[i] = s + ip[i];
    }
}

// ---------------------------------------------------------------------------
extern "C" void kernel_launch(void* const* d_in, const int* in_sizes, int n_in,
                              void* d_out, int out_size, void* d_ws, size_t ws_size,
                              hipStream_t stream)
{
    const float* r = (const float*)d_in[0];
    const float* t = (const float*)d_in[1];
    // params flattened depth-first in dict insertion order
    const float* bnr_g = (const float*)d_in[2];
    const float* bnr_b = (const float*)d_in[3];
    const float* bnr_m = (const float*)d_in[4];
    const float* bnr_v = (const float*)d_in[5];
    const float* bnt_g = (const float*)d_in[6];
    const float* bnt_b = (const float*)d_in[7];
    const float* bnt_m = (const float*)d_in[8];
    const float* bnt_v = (const float*)d_in[9];
    const float* conv_r_w = (const float*)d_in[10];
    const float* conv_r_b = (const float*)d_in[11];
    const float* conv_t_w = (const float*)d_in[12];
    const float* conv_t_b = (const float*)d_in[13];
    const float* act_r_s = (const float*)d_in[14];
    const float* act_r_b = (const float*)d_in[15];
    const float* act_t_s = (const float*)d_in[16];
    const float* act_t_b = (const float*)d_in[17];
    const float* fc1_w = (const float*)d_in[18];
    const float* mlp_s = (const float*)d_in[19];
    const float* mlp_b = (const float*)d_in[20];
    const float* fc2_w = (const float*)d_in[21];
    const float* cw    = (const float*)d_in[22];
    const float* rca_w1 = (const float*)d_in[23];
    const float* rca_b1 = (const float*)d_in[24];
    const float* rca_w2 = (const float*)d_in[25];
    const float* rca_b2 = (const float*)d_in[26];
    const float* rcp_w1 = (const float*)d_in[27];
    const float* rcp_b1 = (const float*)d_in[28];
    const float* rcp_w2 = (const float*)d_in[29];
    const float* rcp_b2 = (const float*)d_in[30];
    const float* tca_w1 = (const float*)d_in[31];
    const float* tca_b1 = (const float*)d_in[32];
    const float* tca_w2 = (const float*)d_in[33];
    const float* tca_b2 = (const float*)d_in[34];
    const float* tcp_w1 = (const float*)d_in[35];
    const float* tcp_b1 = (const float*)d_in[36];
    const float* tcp_w2 = (const float*)d_in[37];
    const float* tcp_b2 = (const float*)d_in[38];
    const float* phr_w1 = (const float*)d_in[39];
    const float* phr_b1 = (const float*)d_in[40];
    const float* phr_w2 = (const float*)d_in[41];
    const float* phr_b2 = (const float*)d_in[42];
    const float* pht_w1 = (const float*)d_in[43];
    const float* pht_b1 = (const float*)d_in[44];
    const float* pht_w2 = (const float*)d_in[45];
    const float* pht_b2 = (const float*)d_in[46];
    const float* ffn_g = (const float*)d_in[47];
    const float* ffn_b = (const float*)d_in[48];
    const float* ffn_m = (const float*)d_in[49];
    const float* ffn_v = (const float*)d_in[50];
    const float* dw_w  = (const float*)d_in[51];

    float* ws = (float*)d_ws;
    size_t off = 0;
    auto alloc = [&](size_t n) { float* p = ws + off; off += n; return p; };
    const size_t PLANE = (size_t)NB * DIMC * HWQ;   // 8.39M
    const size_t FREQ  = (size_t)NB * DIMC * SFSQ;  // 4.33M
    const size_t CFREQ = (size_t)NB * C2Q * SFSQ;   // 2.20M

    float* RES  = alloc(PLANE);
    float* RN   = alloc(PLANE);
    float* TN   = alloc(PLANE);
    float* R2   = alloc(PLANE);
    float* T2   = alloc(PLANE);
    float* OUT1 = alloc(PLANE);
    float* RSA  = alloc(FREQ);
    float* RSP  = alloc(FREQ);
    float* TSA  = alloc(FREQ);
    float* TSP  = alloc(FREQ);
    float* RSC  = alloc(FREQ);
    float* TSC  = alloc(FREQ);
    float* RSPE = alloc(FREQ);
    float* TSPE = alloc(FREQ);
    float* TMP  = alloc(FREQ);
    float* CFA  = alloc(CFREQ);
    float* CFP  = alloc(CFREQ);
    float* AE   = alloc(CFREQ);
    float* PE   = alloc(CFREQ);
    float* POOL = alloc((size_t)NB * DIMC);
    float* ROUTE= alloc((size_t)NB * NFQ * DIMC);
    (void)ws_size; (void)in_sizes; (void)n_in; (void)out_size;

    // 1) BN + res + pooled
    k_bn_res_pool<<<NB * DIMC, 256, 0, stream>>>(r, t, bnr_g, bnr_b, bnr_m, bnr_v,
                                                 bnt_g, bnt_b, bnt_m, bnt_v,
                                                 RN, TN, RES, POOL);
    // 2) routing MLP
    k_routing<<<NB, DIMC, 0, stream>>>(POOL, fc1_w, mlp_s, mlp_b, fc2_w, ROUTE);

    // 3) branch convs + StarReLU (WMMA GEMM, M=K=128, N=4096/batch)
    long pbs = (long)DIMC * HWQ;
    dim3 gconv(HWQ / 64, DIMC / 32, NB);
    k_wmma_gemm<DIMC, DIMC><<<gconv, 128, 0, stream>>>(conv_r_w, RN, conv_r_b, R2, HWQ,
                                                       pbs, pbs, 2, act_r_s, act_r_b);
    k_wmma_gemm<DIMC, DIMC><<<gconv, 128, 0, stream>>>(conv_t_w, TN, conv_t_b, T2, HWQ,
                                                       pbs, pbs, 2, act_t_s, act_t_b);

    // 4) spatial rfft2 -> amp/phase
    k_rfft2_amp_phase<<<NB * DIMC, 256, 0, stream>>>(R2, RSA, RSP);
    k_rfft2_amp_phase<<<NB * DIMC, 256, 0, stream>>>(T2, TSA, TSP);

    // 5-7) channel enhance for r branch
    long cbs = (long)C2Q * SFSQ;
    dim3 gcf(SFSQ / 64, NB);
    dim3 gch(SFSQ / 64, (C2Q + 31) / 32, NB);
    k_chan_rfft<<<gcf, 64, 0, stream>>>(RSA, CFA, CFP);
    k_wmma_gemm<C2Q, C2Q><<<gch, 128, 0, stream>>>(rca_w1, CFA, rca_b1, TMP, SFSQ, cbs, cbs, 1, mlp_s, mlp_b);
    k_wmma_gemm<C2Q, C2Q><<<gch, 128, 0, stream>>>(rca_w2, TMP, rca_b2, AE,  SFSQ, cbs, cbs, 0, mlp_s, mlp_b);
    k_wmma_gemm<C2Q, C2Q><<<gch, 128, 0, stream>>>(rcp_w1, CFP, rcp_b1, TMP, SFSQ, cbs, cbs, 1, mlp_s, mlp_b);
    k_wmma_gemm<C2Q, C2Q><<<gch, 128, 0, stream>>>(rcp_w2, TMP, rcp_b2, PE,  SFSQ, cbs, cbs, 0, mlp_s, mlp_b);
    k_chan_irfft_mul<<<gcf, 64, 0, stream>>>(AE, PE, RSA, RSC);

    // channel enhance for t branch (reuse CF/AE/PE buffers)
    k_chan_rfft<<<gcf, 64, 0, stream>>>(TSA, CFA, CFP);
    k_wmma_gemm<C2Q, C2Q><<<gch, 128, 0, stream>>>(tca_w1, CFA, tca_b1, TMP, SFSQ, cbs, cbs, 1, mlp_s, mlp_b);
    k_wmma_gemm<C2Q, C2Q><<<gch, 128, 0, stream>>>(tca_w2, TMP, tca_b2, AE,  SFSQ, cbs, cbs, 0, mlp_s, mlp_b);
    k_wmma_gemm<C2Q, C2Q><<<gch, 128, 0, stream>>>(tcp_w1, CFP, tcp_b1, TMP, SFSQ, cbs, cbs, 1, mlp_s, mlp_b);
    k_wmma_gemm<C2Q, C2Q><<<gch, 128, 0, stream>>>(tcp_w2, TMP, tcp_b2, PE,  SFSQ, cbs, cbs, 0, mlp_s, mlp_b);
    k_chan_irfft_mul<<<gcf, 64, 0, stream>>>(AE, PE, TSA, TSC);

    // 8) phase enhancers (M=K=128, N=2112/batch)
    long fbs = (long)DIMC * SFSQ;
    dim3 gph(SFSQ / 64, DIMC / 32, NB);
    k_wmma_gemm<DIMC, DIMC><<<gph, 128, 0, stream>>>(phr_w1, RSP, phr_b1, TMP,  SFSQ, fbs, fbs, 1, mlp_s, mlp_b);
    k_wmma_gemm<DIMC, DIMC><<<gph, 128, 0, stream>>>(phr_w2, TMP, phr_b2, RSPE, SFSQ, fbs, fbs, 0, mlp_s, mlp_b);
    k_wmma_gemm<DIMC, DIMC><<<gph, 128, 0, stream>>>(pht_w1, TSP, pht_b1, TMP,  SFSQ, fbs, fbs, 1, mlp_s, mlp_b);
    k_wmma_gemm<DIMC, DIMC><<<gph, 128, 0, stream>>>(pht_w2, TMP, pht_b2, TSPE, SFSQ, fbs, fbs, 0, mlp_s, mlp_b);

    // 9) combine, dynamic complex filter, irfft2, +res
    k_combine_irfft2<<<NB * DIMC, 256, 0, stream>>>(RSC, RSPE, TSC, TSPE, ROUTE, cw, RES, OUT1);

    // 10) DwconvFFN
    k_ffn<<<NB * DIMC, 256, 0, stream>>>(OUT1, ffn_g, ffn_b, ffn_m, ffn_v, dw_w, (float*)d_out);
}
